// MLSTMFixD_57887569216037
// MI455X (gfx1250) — matrix operand order repaired and checked by reading, not compiled, not run on hardware
//
#include <hip/hip_runtime.h>
#include <hip/hip_bf16.h>

// ---------------------------------------------------------------------------
// MLSTM-FixD on gfx1250: persistent single-workgroup WMMA recurrence.
// Round 4: keep GEMM K-loops ROLLED (#pragma unroll 1). Full unrolling made
// the compiler hoist every B load ahead of the barrier and spill to scratch.
// ---------------------------------------------------------------------------

typedef __attribute__((ext_vector_type(16))) __bf16 v16bf;
typedef __attribute__((ext_vector_type(8)))  float  v8f;

#define WMMA_BF16(a,b,c) __builtin_amdgcn_wmma_f32_16x16x32_bf16( \
    false,(a),false,(b),(short)0,(c),false,false)

#if defined(__has_builtin)
#if __has_builtin(__builtin_amdgcn_sched_barrier)
#define SCHED_FENCE() __builtin_amdgcn_sched_barrier(0)
#endif
#endif
#ifndef SCHED_FENCE
#define SCHED_FENCE()
#endif

__device__ __forceinline__ float sigm(float x){ return 1.f/(1.f+__expf(-x)); }
__device__ __forceinline__ float tanh_f(float x){ return 2.f/(1.f+__expf(-2.f*x)) - 1.f; }
__device__ __forceinline__ unsigned short bf16b(float x){
  __bf16 h=(__bf16)x; return __builtin_bit_cast(unsigned short,h);
}
// Inverse of the 16-bit A-matrix fragment layout (contiguous k = 16*half + elem, matching B packing):
// given row-in-tile ml (0..15) and k-in-tile kl (0..31), produce (lane, elem).
__device__ __forceinline__ void a_lane_elem(int ml,int kl,int&l,int&e){
  int hi=kl>>4; l=ml+(hi<<4); e=kl&15;
}

// ---------------- workspace layout (bytes) ----------------
constexpr size_t A256(size_t x){ return (x+255)&~(size_t)255; }
constexpr size_t O_WC  = 0;                                  // Wc = Wih1@Who0  (1536x512 f32)
constexpr size_t O_BC  = A256(O_WC + (size_t)1536*512*4);    // bc (1536 f32)
constexpr size_t O_WD  = A256(O_BC + 1536*4);                // wd (32x512 f32)
constexpr size_t O_W0  = A256(O_WD + (size_t)32*512*4);      // W0cat packed bf16 (96nt x20kt x512)
constexpr size_t O_W1  = A256(O_W0 + (size_t)983040*2);      // W1cat packed bf16 (96nt x32kt x512)
constexpr size_t O_WO  = A256(O_W1 + (size_t)1572864*2);     // Who1 packed bf16 (4nt x16kt x512)
constexpr size_t O_EMB = A256(O_WO + (size_t)32768*2);       // emb fragments (256t x 4096)
constexpr size_t O_R0  = A256(O_EMB + (size_t)1048576*2);    // ring0 (32x32x512 f32)
constexpr size_t O_R1  = A256(O_R0 + (size_t)524288*4);      // ring1
constexpr size_t O_H1  = A256(O_R1 + (size_t)524288*4);      // h1 fragments (512mt x16kt x512)
constexpr int SMEM_USHORTS = 20*2*512 + 32*2*512;            // af0 + af1 = 53248
constexpr int SMEM_BYTES   = SMEM_USHORTS*2;                 // 106496

// ---------------- prep: Wc = Wih1 @ Who0 ----------------
__global__ void k_prep_wc(const float* __restrict__ Wih1,
                          const float* __restrict__ Who0,
                          float* __restrict__ wc){
  int idx = blockIdx.x*256 + threadIdx.x;
  if (idx >= 1536*512) return;
  int n = idx>>9, k = idx&511;
  float s = 0.f;
  for (int j=0;j<512;++j) s += Wih1[n*512+j]*Who0[j*512+k];
  wc[idx] = s;
}

// ---------------- prep: bc and fractional weights wd ----------------
__global__ void k_prep_small(const float* __restrict__ Wih1,
                             const float* __restrict__ bho0,
                             const float* __restrict__ b1,
                             const float* __restrict__ dvals,
                             float* __restrict__ bcO,
                             float* __restrict__ wdO){
  int i = blockIdx.x*256 + threadIdx.x;
  if (i < 1536){
    float s = b1[i];
    for (int j=0;j<512;++j) s += Wih1[i*512+j]*bho0[j];
    bcO[i] = s;
  } else if (i < 2048){
    int h = i - 1536;
    float d = 0.5f*sigm(dvals[h]);
    float a = 1.f;
    for (int j=1;j<=32;++j){                 // a_j = a_{j-1}*(j-1-d)/j
      a *= ((float)(j-1) - d)/(float)j;
      wdO[(32-j)*512 + h] = a;               // wd[0]=a_32 (lag 32) ... wd[31]=a_1
    }
  }
}

// ---------------- prep: pack weights into B-matrix fragment order ----------
// B 32x16 bf16 packing used here (contiguous k, same as A): lane<16 -> K=e, lane>=16 -> K=16+e; N = lane&15.
__global__ void k_pack(const float* __restrict__ Wih0,const float* __restrict__ Whh0,
                       const float* __restrict__ wc,  const float* __restrict__ Whh1,
                       const float* __restrict__ Who1,
                       unsigned short* __restrict__ w0p,
                       unsigned short* __restrict__ w1p,
                       unsigned short* __restrict__ wop){
  const int S0=983040, S1=1572864, S2=32768;
  for (int i = blockIdx.x*256+threadIdx.x; i < S0+S1+S2; i += gridDim.x*256){
    if (i < S0){                                      // W0cat: K = [Wih0(128) | Whh0(512)]
      int e=i&15, l=(i>>4)&31, rest=i>>9;
      int kt=rest%20, nt=rest/20;
      int n=nt*16+(l&15), k=kt*32+e+((l>>4)<<4);
      float v = (k<128) ? Wih0[n*128+k] : Whh0[n*512+(k-128)];
      w0p[i] = bf16b(v);
    } else if (i < S0+S1){                            // W1cat: K = [Wc(512) | Whh1(512)]
      int j=i-S0;
      int e=j&15, l=(j>>4)&31, rest=j>>9;
      int kt=rest&31, nt=rest>>5;
      int n=nt*16+(l&15), k=kt*32+e+((l>>4)<<4);
      float v = (k<512) ? wc[n*512+k] : Whh1[n*512+(k-512)];
      w1p[j] = bf16b(v);
    } else {                                          // Who1 (64x512)
      int j=i-S0-S1;
      int e=j&15, l=(j>>4)&31, rest=j>>9;
      int kt=rest&15, nt=rest>>4;
      int n=nt*16+(l&15), k=kt*32+e+((l>>4)<<4);
      wop[j] = bf16b(Who1[n*512+k]);
    }
  }
}

// ---------------- prep: embedding GEMM -> A-fragment layout per timestep ---
__global__ void k_emb(const float* __restrict__ inputs,
                      const float* __restrict__ W_emb,
                      const float* __restrict__ b_emb,
                      unsigned short* __restrict__ embf){
  int idx = blockIdx.x*256 + threadIdx.x;
  if (idx >= 256*32*128) return;
  int col = idx&127, b=(idx>>7)&31, t=idx>>12;
  float s = b_emb[col];
  const float* xp = inputs + (t*32+b)*32;
  const float* wp = W_emb + col*32;
  for (int j=0;j<32;++j) s += xp[j]*wp[j];
  int l,e; a_lane_elem(b&15, col&31, l, e);
  embf[t*4096 + (((col>>5)*2 + (b>>4))*32 + l)*16 + e] = bf16b(s);
}

__global__ void k_zero(float* __restrict__ p, int n){
  int i = blockIdx.x*256 + threadIdx.x;
  if (i < n) p[i] = 0.f;
}

// ---------------- main persistent recurrence kernel ------------------------
// 1 workgroup, 512 threads = 16 waves (4/SIMD -> ~256 VGPRs/wave).
// Wave w owns hidden columns h in [32w, 32w+32) as two h-groups u=0,1;
// for each h-group the gate tiles are nt = {hg, hg+32, hg+64} = {i,o,g}.
__global__ __launch_bounds__(512) void k_main(
    const unsigned short* __restrict__ w0p, const unsigned short* __restrict__ w1p,
    const float* __restrict__ b0, const float* __restrict__ bc,
    const float* __restrict__ wd,
    const unsigned short* __restrict__ embf,
    float* __restrict__ ring0, float* __restrict__ ring1,
    unsigned short* __restrict__ h1f)
{
  extern __shared__ unsigned short smem[];
  unsigned short* af0 = smem;              // cell0 A frags: [kt(20)][mt(2)][lane32][16]
  unsigned short* af1 = smem + 20*2*512;   // cell1 A frags: [kt(32)][mt(2)][lane32][16]
  const int tid=threadIdx.x, wave=tid>>5, lane=tid&31;
  const int lan=lane&15, hif=lane>>4;

  for (int i=tid;i<SMEM_USHORTS;i+=512) smem[i]=0;   // h0=h1=0 initial state
  __syncthreads();

  // per-(u,g) weight base pointers for this wave/lane (advance by 512 elems/kt)
  const unsigned short* w0base[2][3];
  const unsigned short* w1base[2][3];
  #pragma unroll
  for (int u=0;u<2;++u)
    #pragma unroll
    for (int g=0;g<3;++g){
      const int nt = (wave*2+u) + 32*g;
      w0base[u][g] = w0p + (nt*20*32 + lane)*16;
      w1base[u][g] = w1p + (nt*32*32 + lane)*16;
    }

  for (int t=0;t<256;++t){
    // stage emb_t into cell0 A fragments (K 0..127)
    {
      const unsigned short* src = embf + t*4096;
      #pragma unroll 1
      for (int i=tid;i<4096;i+=512) af0[i]=src[i];
    }
    __syncthreads();

    v8f acc[2][3][2];
    // ------------- cell 0: z = [emb_t | h0] @ W0cat^T + b0 -------------
    #pragma unroll
    for (int u=0;u<2;++u){
      const int hc = (wave*2+u)*16 + lan;
      #pragma unroll
      for (int g=0;g<3;++g){
        float bv = b0[g*512 + hc];
        v8f a;
        #pragma unroll
        for (int r=0;r<8;++r) a[r]=bv;
        acc[u][g][0]=a; acc[u][g][1]=a;
      }
    }
    #pragma unroll 1
    for (int kt=0;kt<20;++kt){
      v16bf a0 = *(const v16bf*)(af0 + ((kt*2+0)*32+lane)*16);
      v16bf a1 = *(const v16bf*)(af0 + ((kt*2+1)*32+lane)*16);
      #pragma unroll
      for (int u=0;u<2;++u){
        #pragma unroll
        for (int g=0;g<3;++g){
          v16bf bb = *(const v16bf*)(w0base[u][g] + kt*512);
          acc[u][g][0]=WMMA_BF16(a0,bb,acc[u][g][0]);
          acc[u][g][1]=WMMA_BF16(a1,bb,acc[u][g][1]);
        }
      }
      SCHED_FENCE();   // no cross-iteration pipelining
    }
    __syncthreads();                       // all waves done reading af0
    // ------------- cell 0 epilogue: gates + ARFIMA sum, in registers ----
    #pragma unroll
    for (int u=0;u<2;++u){
      const int hc = (wave*2+u)*16 + lan;
      float wdv[32];
      #pragma unroll
      for (int k=0;k<32;++k) wdv[k]=wd[k*512+hc];
      #pragma unroll
      for (int mt=0;mt<2;++mt){
        #pragma unroll
        for (int r=0;r<8;++r){
          const int brow = mt*16 + hif*8 + r;            // batch row (C layout)
          float frac=0.f;
          #pragma unroll
          for (int k=0;k<32;++k)
            frac += wdv[k]*ring0[((((t+k)&31)*32+brow)<<9)+hc];
          const float c = sigm(acc[u][0][mt][r])*tanh_f(acc[u][2][mt][r]) - frac;
          ring0[(((t&31)*32+brow)<<9)+hc] = c;
          const unsigned short hb = bf16b(sigm(acc[u][1][mt][r])*tanh_f(c));
          int l,e; a_lane_elem(hif*8+r, hc&31, l, e);
          const int ktA = hc>>5;
          af1[((ktA*2+mt)*32+l)*16+e]       = hb;        // cell1 A, K 0..511
          af0[(((ktA+4)*2+mt)*32+l)*16+e]   = hb;        // cell0 next step, K 128..639
        }
      }
    }
    __syncthreads();
    // ------------- cell 1: z = [h0_new | h1] @ W1cat^T + bc -------------
    #pragma unroll
    for (int u=0;u<2;++u){
      const int hc = (wave*2+u)*16 + lan;
      #pragma unroll
      for (int g=0;g<3;++g){
        float bv = bc[g*512 + hc];
        v8f a;
        #pragma unroll
        for (int r=0;r<8;++r) a[r]=bv;
        acc[u][g][0]=a; acc[u][g][1]=a;
      }
    }
    #pragma unroll 1
    for (int kt=0;kt<32;++kt){
      v16bf a0 = *(const v16bf*)(af1 + ((kt*2+0)*32+lane)*16);
      v16bf a1 = *(const v16bf*)(af1 + ((kt*2+1)*32+lane)*16);
      #pragma unroll
      for (int u=0;u<2;++u){
        #pragma unroll
        for (int g=0;g<3;++g){
          v16bf bb = *(const v16bf*)(w1base[u][g] + kt*512);
          acc[u][g][0]=WMMA_BF16(a0,bb,acc[u][g][0]);
          acc[u][g][1]=WMMA_BF16(a1,bb,acc[u][g][1]);
        }
      }
      SCHED_FENCE();   // no cross-iteration pipelining
    }
    __syncthreads();
    // ------------- cell 1 epilogue -------------
    #pragma unroll
    for (int u=0;u<2;++u){
      const int hc = (wave*2+u)*16 + lan;
      float wdv[32];
      #pragma unroll
      for (int k=0;k<32;++k) wdv[k]=wd[k*512+hc];
      #pragma unroll
      for (int mt=0;mt<2;++mt){
        #pragma unroll
        for (int r=0;r<8;++r){
          const int brow = mt*16 + hif*8 + r;
          float frac=0.f;
          #pragma unroll
          for (int k=0;k<32;++k)
            frac += wdv[k]*ring1[((((t+k)&31)*32+brow)<<9)+hc];
          const float c = sigm(acc[u][0][mt][r])*tanh_f(acc[u][2][mt][r]) - frac;
          ring1[(((t&31)*32+brow)<<9)+hc] = c;
          const unsigned short hb = bf16b(sigm(acc[u][1][mt][r])*tanh_f(c));
          int l,e; a_lane_elem(hif*8+r, hc&31, l, e);
          const int ktA = hc>>5;
          af1[(((ktA+16)*2+mt)*32+l)*16+e] = hb;           // cell1 next step, K 512..1023
          h1f[(((t*2+mt)*16+ktA)*32+l)*16+e] = hb;         // batched output GEMM input
        }
      }
    }
    __syncthreads();
  }
}

// ---------------- batched output GEMM: out = H1 @ Who1^T + bho1 ------------
// M = 8192 (T*B), N = 64, K = 512. 2048 tiles; 256 blocks x 8 waves.
__global__ __launch_bounds__(256) void k_out(const unsigned short* __restrict__ h1f,
                                             const unsigned short* __restrict__ wop,
                                             const float* __restrict__ bho1,
                                             float* __restrict__ out){
  const int wave = threadIdx.x>>5, lane = threadIdx.x&31;
  const int tile = blockIdx.x*8 + wave;
  const int mt = tile>>2, nt = tile&3;
  const float bv = bho1[nt*16 + (lane&15)];
  v8f acc;
  #pragma unroll
  for (int r=0;r<8;++r) acc[r]=bv;
  #pragma unroll 1
  for (int kt=0;kt<16;++kt){
    v16bf a = *(const v16bf*)(h1f + (((size_t)mt*16+kt)*32+lane)*16);
    v16bf b = *(const v16bf*)(wop + ((nt*16+kt)*32+lane)*16);
    acc = WMMA_BF16(a,b,acc);
  }
  const int col = nt*16 + (lane&15);
  #pragma unroll
  for (int r=0;r<8;++r){
    const int row = mt*16 + (lane>>4)*8 + r;
    out[row*64 + col] = acc[r];
  }
}

// ---------------------------------------------------------------------------
extern "C" void kernel_launch(void* const* d_in, const int* in_sizes, int n_in,
                              void* d_out, int out_size, void* d_ws, size_t ws_size,
                              hipStream_t stream) {
  (void)in_sizes; (void)n_in; (void)out_size; (void)ws_size;
  const float* inputs=(const float*)d_in[0];
  const float* W_emb =(const float*)d_in[1];
  const float* b_emb =(const float*)d_in[2];
  const float* dvals =(const float*)d_in[3];
  const float* Wih0  =(const float*)d_in[4];
  const float* Whh0  =(const float*)d_in[5];
  const float* b0    =(const float*)d_in[6];
  const float* Who0  =(const float*)d_in[7];
  const float* bho0  =(const float*)d_in[8];
  const float* Wih1  =(const float*)d_in[9];
  const float* Whh1  =(const float*)d_in[10];
  const float* b1    =(const float*)d_in[11];
  const float* Who1  =(const float*)d_in[12];
  const float* bho1  =(const float*)d_in[13];

  char* ws=(char*)d_ws;
  float* wc   =(float*)(ws+O_WC);
  float* bc   =(float*)(ws+O_BC);
  float* wd   =(float*)(ws+O_WD);
  unsigned short* w0p =(unsigned short*)(ws+O_W0);
  unsigned short* w1p =(unsigned short*)(ws+O_W1);
  unsigned short* wop =(unsigned short*)(ws+O_WO);
  unsigned short* embf=(unsigned short*)(ws+O_EMB);
  float* ring0=(float*)(ws+O_R0);
  float* ring1=(float*)(ws+O_R1);
  unsigned short* h1f =(unsigned short*)(ws+O_H1);
  float* out=(float*)d_out;

  (void)hipFuncSetAttribute((const void*)k_main,
        hipFuncAttributeMaxDynamicSharedMemorySize, SMEM_BYTES);

  k_prep_wc  <<<3072,256,0,stream>>>(Wih1,Who0,wc);
  k_prep_small<<<8,256,0,stream>>>(Wih1,bho0,b1,dvals,bc,wd);
  k_pack     <<<4096,256,0,stream>>>(Wih0,Whh0,wc,Whh1,Who1,w0p,w1p,wop);
  k_emb      <<<4096,256,0,stream>>>(inputs,W_emb,b_emb,embf);
  k_zero     <<<4096,256,0,stream>>>(ring0, 1048576);   // ring0+ring1 contiguous
  k_main     <<<1,512,SMEM_BYTES,stream>>>(w0p,w1p,b0,bc,wd,embf,ring0,ring1,h1f);
  k_out      <<<256,256,0,stream>>>(h1f,wop,bho1,out);
  (void)ring1;
}
